// SensorModulator_29721173688725
// MI455X (gfx1250) — matrix operand order repaired and checked by reading, not compiled, not run on hardware
//
#include <hip/hip_runtime.h>
#include <hip/hip_bf16.h>
#include <math.h>

// ---------------------------------------------------------------------------
// Model constants (match reference)
// ---------------------------------------------------------------------------
#define BB   32
#define TT   2048
#define NT   (BB * TT)     // 65536 rows
#define HID  256
#define DIN  512
#define DSTATE 16
#define DTRANK 16
#define DCONV  4
#define DM   1024

typedef __attribute__((ext_vector_type(16))) __bf16 v16bf;
typedef __attribute__((ext_vector_type(8)))  __bf16 v8bf;
typedef __attribute__((ext_vector_type(8)))  float  v8f;

// ---------------------------------------------------------------------------
// Embedding: x[bt,h] = sensor[bt,:6] . emb_w[h,:6] + emb_b[h]
// ---------------------------------------------------------------------------
__global__ __launch_bounds__(256)
void embed_kernel(const float* __restrict__ sensor,
                  const float* __restrict__ emb_w,
                  const float* __restrict__ emb_b,
                  float* __restrict__ x)
{
    int i = blockIdx.x * blockDim.x + threadIdx.x;     // over NT*HID
    if (i >= NT * HID) return;
    int h  = i & (HID - 1);
    int bt = i >> 8;
    const float* s = sensor + (size_t)bt * 6;
    float acc = emb_b[h];
#pragma unroll
    for (int k = 0; k < 6; ++k) acc += s[k] * emb_w[h * 6 + k];
    x[i] = acc;
}

// ---------------------------------------------------------------------------
// LayerNorm over HID=256, one wave32 per row, 8 rows per 256-thread block
// ---------------------------------------------------------------------------
__global__ __launch_bounds__(256)
void ln_kernel(const float* __restrict__ x,
               const float* __restrict__ w,
               const float* __restrict__ b,
               float* __restrict__ y, int nrows)
{
    int gid  = blockIdx.x * blockDim.x + threadIdx.x;
    int row  = gid >> 5;
    int lane = threadIdx.x & 31;
    if (row >= nrows) return;
    const float* rp = x + (size_t)row * HID;
    float v[8];
    float s = 0.f;
#pragma unroll
    for (int i = 0; i < 8; ++i) { v[i] = rp[lane + 32 * i]; s += v[i]; }
#pragma unroll
    for (int m = 16; m > 0; m >>= 1) s += __shfl_xor(s, m, 32);
    float mean = s * (1.0f / HID);
    float vs = 0.f;
#pragma unroll
    for (int i = 0; i < 8; ++i) { float d = v[i] - mean; vs += d * d; }
#pragma unroll
    for (int m = 16; m > 0; m >>= 1) vs += __shfl_xor(vs, m, 32);
    float inv = rsqrtf(vs * (1.0f / HID) + 1e-5f);
    float* yp = y + (size_t)row * HID;
#pragma unroll
    for (int i = 0; i < 8; ++i) {
        int c = lane + 32 * i;
        yp[c] = (v[i] - mean) * inv * w[c] + b[c];
    }
}

// ---------------------------------------------------------------------------
// Generic GEMM: Out[M,N] = A[M,K] * W[N,K]^T (+ Res), fp32 in/out,
// bf16 WMMA compute.
//   Block: 256 threads (8 waves), tile 256(M) x 64(N), K-step 32.
//   Each wave owns a 32x64 sub-tile -> 8 WMMAs / K-step.
//   Register double-buffered LDS staging: global loads for K-step k+32
//   are issued before computing K-step k; one __syncthreads per K-step.
//   Requires M % 256 == 0 and K % 32 == 0 (true for all launches here).
//   LDS stride 40 halfwords (80B) -> conflict-free ds_load_b128 fragments.
// ---------------------------------------------------------------------------
#define LDS_STRIDE 40

__global__ __launch_bounds__(256)
void gemm_bf16_wmma(const float* __restrict__ A,
                    const float* __restrict__ W,
                    const float* __restrict__ Res,
                    float* __restrict__ Out,
                    int M, int N, int K, int addRes)
{
    __shared__ __bf16 sA[2][256 * LDS_STRIDE];
    __shared__ __bf16 sB[2][64 * LDS_STRIDE];

    const int m0   = blockIdx.x * 256;
    const int n0   = blockIdx.y * 64;
    const int tid  = threadIdx.x;
    const int lane = tid & 31;
    const int wave = tid >> 5;

    float4 ra[8];          // staged A quads (256x32 / 256 threads)
    float4 rb[2];          // staged W quads (64x32 / 256 threads)

    auto load_tiles = [&](int k0) {
#pragma unroll
        for (int i = 0; i < 8; ++i) {
            int q  = tid + i * 256;
            int r  = q >> 3;
            int cg = (q & 7) << 2;
            ra[i] = *(const float4*)(A + (size_t)(m0 + r) * K + k0 + cg);
        }
#pragma unroll
        for (int i = 0; i < 2; ++i) {
            int q  = tid + i * 256;
            int r  = q >> 3;
            int cg = (q & 7) << 2;
            int gn = n0 + r;
            rb[i] = (gn < N) ? *(const float4*)(W + (size_t)gn * K + k0 + cg)
                             : make_float4(0.f, 0.f, 0.f, 0.f);
        }
    };

    auto stage_lds = [&](int buf) {
#pragma unroll
        for (int i = 0; i < 8; ++i) {
            int q  = tid + i * 256;
            int r  = q >> 3;
            int cg = (q & 7) << 2;
            __bf16* p = &sA[buf][r * LDS_STRIDE + cg];
            p[0] = (__bf16)ra[i].x; p[1] = (__bf16)ra[i].y;
            p[2] = (__bf16)ra[i].z; p[3] = (__bf16)ra[i].w;
        }
#pragma unroll
        for (int i = 0; i < 2; ++i) {
            int q  = tid + i * 256;
            int r  = q >> 3;
            int cg = (q & 7) << 2;
            __bf16* p = &sB[buf][r * LDS_STRIDE + cg];
            p[0] = (__bf16)rb[i].x; p[1] = (__bf16)rb[i].y;
            p[2] = (__bf16)rb[i].z; p[3] = (__bf16)rb[i].w;
        }
    };

    v8f acc[2][4] = {};

    const int fr    = lane & 15;
    const int khalf = (lane >> 4) << 3;   // A frag: 0 or 8   (split-K layout)
    const int kbase = (lane >> 4) << 4;   // unused; B frag uses khalf (split-K layout)

    auto compute = [&](int buf) {
        v16bf af[2];
#pragma unroll
        for (int s = 0; s < 2; ++s) {
            const __bf16* base = &sA[buf][(wave * 32 + s * 16 + fr) * LDS_STRIDE];
            v8bf lo = *(const v8bf*)(base + khalf);
            v8bf hi = *(const v8bf*)(base + khalf + 16);
#pragma unroll
            for (int i = 0; i < 8; ++i) { af[s][i] = lo[i]; af[s][i + 8] = hi[i]; }
        }
#pragma unroll
        for (int ns = 0; ns < 4; ++ns) {
            const __bf16* bb = &sB[buf][(ns * 16 + fr) * LDS_STRIDE];
            v8bf b0 = *(const v8bf*)(bb + khalf);
            v8bf b1 = *(const v8bf*)(bb + khalf + 16);
            v16bf bv;
#pragma unroll
            for (int i = 0; i < 8; ++i) { bv[i] = b0[i]; bv[i + 8] = b1[i]; }
            acc[0][ns] = __builtin_amdgcn_wmma_f32_16x16x32_bf16(
                false, af[0], false, bv, (short)0, acc[0][ns], false, false);
            acc[1][ns] = __builtin_amdgcn_wmma_f32_16x16x32_bf16(
                false, af[1], false, bv, (short)0, acc[1][ns], false, false);
        }
    };

    // Prologue: stage K-step 0
    load_tiles(0);
    stage_lds(0);
    __syncthreads();

    int buf = 0;
    for (int k0 = 0; k0 < K; k0 += 32) {
        const bool more = (k0 + 32) < K;
        if (more) load_tiles(k0 + 32);   // global loads overlap WMMA below
        compute(buf);
        if (more) {
            stage_lds(buf ^ 1);
            __syncthreads();
            buf ^= 1;
        }
    }

    // Store: D element g, lane L -> row M = g + (L>=16 ? 8:0), col N = L&15
    const int mh = (lane >> 4) << 3;
#pragma unroll
    for (int s = 0; s < 2; ++s) {
#pragma unroll
        for (int ns = 0; ns < 4; ++ns) {
            int col = n0 + ns * 16 + fr;
            if (col >= N) continue;
#pragma unroll
            for (int g = 0; g < 8; ++g) {
                int row = m0 + wave * 32 + s * 16 + g + mh;
                size_t o = (size_t)row * N + col;
                float v = acc[s][ns][g];
                if (addRes) v += Res[o];
                Out[o] = v;
            }
        }
    }
}

// ---------------------------------------------------------------------------
// Depthwise causal conv (DCONV=4) + bias + SiLU.  xin = xz[...,0:512]
// ---------------------------------------------------------------------------
__global__ __launch_bounds__(256)
void conv_silu_kernel(const float* __restrict__ xz,
                      const float* __restrict__ cw,
                      const float* __restrict__ cb,
                      float* __restrict__ u)
{
    int i = blockIdx.x * blockDim.x + threadIdx.x;     // over NT*DIN
    if (i >= NT * DIN) return;
    int d  = i & (DIN - 1);
    int bt = i >> 9;
    int t  = bt & (TT - 1);
    int b  = bt >> 11;
    float acc = cb[d];
#pragma unroll
    for (int k = 0; k < DCONV; ++k) {
        int tt = t - (DCONV - 1) + k;
        if (tt >= 0)
            acc += xz[((size_t)(b * TT + tt)) * 1024 + d] * cw[d * DCONV + k];
    }
    u[i] = acc / (1.0f + __expf(-acc));                // silu
}

// ---------------------------------------------------------------------------
// delta = softplus(dt @ dt_w^T + dt_b); dt = xdbl[:, :16]
// ---------------------------------------------------------------------------
__global__ __launch_bounds__(256)
void delta_kernel(const float* __restrict__ xdbl,
                  const float* __restrict__ dt_w,
                  const float* __restrict__ dt_b,
                  float* __restrict__ delta)
{
    int i = blockIdx.x * blockDim.x + threadIdx.x;     // over NT*DIN
    if (i >= NT * DIN) return;
    int d  = i & (DIN - 1);
    int bt = i >> 9;
    const float* dtv = xdbl + (size_t)bt * 48;
    const float* wr  = dt_w + d * DTRANK;
    float acc = dt_b[d];
#pragma unroll
    for (int r = 0; r < DTRANK; ++r) acc += dtv[r] * wr[r];
    delta[i] = (acc > 20.f) ? acc : log1pf(__expf(acc));
}

// ---------------------------------------------------------------------------
// Selective scan, fused with D-skip and silu(z) gating.
// One thread per (b,d); 64-timestep LDS staging of shared B/C vectors.
// ---------------------------------------------------------------------------
__global__ __launch_bounds__(256)
void scan_kernel(const float* __restrict__ delta,
                 const float* __restrict__ u,
                 const float* __restrict__ xdbl,
                 const float* __restrict__ xz,
                 const float* __restrict__ A_log,
                 const float* __restrict__ Dp,
                 float* __restrict__ yg)
{
    int b = blockIdx.x >> 1;
    int d = ((blockIdx.x & 1) << 8) + threadIdx.x;

    float A[DSTATE], h[DSTATE];
#pragma unroll
    for (int n = 0; n < DSTATE; ++n) {
        A[n] = -__expf(A_log[d * DSTATE + n]);
        h[n] = 0.f;
    }
    float Dd = Dp[d];

    __shared__ float sBC[64 * 32];    // 64 timesteps x (16 B + 16 C)

    for (int t0 = 0; t0 < TT; t0 += 64) {
        __syncthreads();
        for (int i = threadIdx.x; i < 64 * 32; i += 256) {
            int tt = i >> 5, j = i & 31;
            sBC[i] = xdbl[(size_t)(b * TT + t0 + tt) * 48 + 16 + j];
        }
        __syncthreads();
        for (int tt = 0; tt < 64; ++tt) {
            size_t bt = (size_t)b * TT + t0 + tt;
            float dl = delta[bt * DIN + d];
            float ut = u[bt * DIN + d];
            float du = dl * ut;
            const float* Bc = &sBC[tt * 32];
            float y = 0.f;
#pragma unroll
            for (int n = 0; n < DSTATE; ++n) {
                float dA = __expf(dl * A[n]);
                h[n] = h[n] * dA + du * Bc[n];
                y += h[n] * Bc[16 + n];
            }
            float z = xz[bt * 1024 + DIN + d];
            float g = z / (1.0f + __expf(-z));
            yg[bt * DIN + d] = (y + ut * Dd) * g;
        }
    }
}

// ---------------------------------------------------------------------------
// Final LN of last timestep only -> ctx[b,256].  One wave per batch row.
// ---------------------------------------------------------------------------
__global__ __launch_bounds__(32)
void ctx_ln_kernel(const float* __restrict__ x,
                   const float* __restrict__ w,
                   const float* __restrict__ b,
                   float* __restrict__ ctx)
{
    int lane = threadIdx.x;
    const float* rp = x + ((size_t)blockIdx.x * TT + (TT - 1)) * HID;
    float v[8];
    float s = 0.f;
#pragma unroll
    for (int i = 0; i < 8; ++i) { v[i] = rp[lane + 32 * i]; s += v[i]; }
#pragma unroll
    for (int m = 16; m > 0; m >>= 1) s += __shfl_xor(s, m, 32);
    float mean = s * (1.0f / HID);
    float vs = 0.f;
#pragma unroll
    for (int i = 0; i < 8; ++i) { float dd = v[i] - mean; vs += dd * dd; }
#pragma unroll
    for (int m = 16; m > 0; m >>= 1) vs += __shfl_xor(vs, m, 32);
    float inv = rsqrtf(vs * (1.0f / HID) + 1e-5f);
    float* cp = ctx + (size_t)blockIdx.x * HID;
#pragma unroll
    for (int i = 0; i < 8; ++i) {
        int c = lane + 32 * i;
        cp[c] = (v[i] - mean) * inv * w[c] + b[c];
    }
}

// ---------------------------------------------------------------------------
// Head: mod = ctx @ head_w^T + head_b; gamma = mod[:, :DM]+1 ; beta = rest.
// d_out layout: [gamma (32*1024) | beta (32*1024)]
// ---------------------------------------------------------------------------
__global__ __launch_bounds__(256)
void head_kernel(const float* __restrict__ ctx,
                 const float* __restrict__ hw,
                 const float* __restrict__ hb,
                 float* __restrict__ out)
{
    int i = blockIdx.x * blockDim.x + threadIdx.x;     // over 32*2048
    if (i >= BB * 2 * DM) return;
    int o = i & (2 * DM - 1);
    int b = i >> 11;
    const float* c  = ctx + (size_t)b * HID;
    const float* wr = hw + (size_t)o * HID;
    float acc = hb[o];
#pragma unroll 8
    for (int k = 0; k < HID; ++k) acc += c[k] * wr[k];
    if (o < DM) out[(size_t)b * DM + o] = acc + 1.0f;
    else        out[(size_t)BB * DM + (size_t)b * DM + (o - DM)] = acc;
}

// ---------------------------------------------------------------------------
// Host orchestration
// ---------------------------------------------------------------------------
extern "C" void kernel_launch(void* const* d_in, const int* in_sizes, int n_in,
                              void* d_out, int out_size, void* d_ws, size_t ws_size,
                              hipStream_t stream)
{
    (void)in_sizes; (void)n_in; (void)out_size; (void)ws_size;

    const float* sensor = (const float*)d_in[0];
    const float* emb_w  = (const float*)d_in[1];
    const float* emb_b  = (const float*)d_in[2];
    // layers: 11 tensors each, starting at index 3
    const float* norm_w = (const float*)d_in[25];
    const float* norm_b = (const float*)d_in[26];
    const float* head_w = (const float*)d_in[27];
    const float* head_b = (const float*)d_in[28];
    float* out = (float*)d_out;

    // Workspace layout (floats)
    float* x    = (float*)d_ws;                        // NT*HID
    float* xn   = x    + (size_t)NT * HID;             // NT*HID
    float* xz   = xn   + (size_t)NT * HID;             // NT*1024
    float* u    = xz   + (size_t)NT * 1024;            // NT*DIN
    float* xdbl = u    + (size_t)NT * DIN;             // NT*48
    float* dl   = xdbl + (size_t)NT * 48;              // NT*DIN
    float* yg   = dl   + (size_t)NT * DIN;             // NT*DIN
    float* ctx  = yg   + (size_t)NT * DIN;             // BB*HID

    // 1. Embedding
    embed_kernel<<<(NT * HID) / 256, 256, 0, stream>>>(sensor, emb_w, emb_b, x);

    // 2. Mamba layers
    for (int L = 0; L < 2; ++L) {
        int base = 3 + L * 11;
        const float* ln_w     = (const float*)d_in[base + 0];
        const float* ln_b     = (const float*)d_in[base + 1];
        const float* in_proj  = (const float*)d_in[base + 2];
        const float* conv_w   = (const float*)d_in[base + 3];
        const float* conv_b   = (const float*)d_in[base + 4];
        const float* x_proj   = (const float*)d_in[base + 5];
        const float* dt_w     = (const float*)d_in[base + 6];
        const float* dt_b     = (const float*)d_in[base + 7];
        const float* A_log    = (const float*)d_in[base + 8];
        const float* Dp       = (const float*)d_in[base + 9];
        const float* out_proj = (const float*)d_in[base + 10];

        // LayerNorm
        ln_kernel<<<NT / 8, 256, 0, stream>>>(x, ln_w, ln_b, xn, NT);

        // in_proj: xz = xn @ in_proj^T   (M=NT, N=1024, K=256)
        gemm_bf16_wmma<<<dim3(NT / 256, 1024 / 64), 256, 0, stream>>>(
            xn, in_proj, nullptr, xz, NT, 1024, HID, 0);

        // causal depthwise conv + silu -> u
        conv_silu_kernel<<<(NT * DIN) / 256, 256, 0, stream>>>(xz, conv_w, conv_b, u);

        // x_proj: xdbl = u @ x_proj^T   (M=NT, N=48, K=512)
        gemm_bf16_wmma<<<dim3(NT / 256, 1), 256, 0, stream>>>(
            u, x_proj, nullptr, xdbl, NT, 48, DIN, 0);

        // delta = softplus(dt @ dt_w^T + dt_b)
        delta_kernel<<<(NT * DIN) / 256, 256, 0, stream>>>(xdbl, dt_w, dt_b, dl);

        // selective scan + gating -> yg
        scan_kernel<<<BB * (DIN / 256), 256, 0, stream>>>(dl, u, xdbl, xz, A_log, Dp, yg);

        // out_proj with residual: x = x + yg @ out_proj^T  (M=NT, N=256, K=512)
        gemm_bf16_wmma<<<dim3(NT / 256, HID / 64), 256, 0, stream>>>(
            yg, out_proj, x, x, NT, HID, DIN, 1);
    }

    // 3. Final LN of last timestep + head
    ctx_ln_kernel<<<BB, 32, 0, stream>>>(x, norm_w, norm_b, ctx);
    head_kernel<<<(BB * 2 * DM) / 256, 256, 0, stream>>>(ctx, head_w, head_b, out);
}
